// Network_13176959664510
// MI455X (gfx1250) — matrix operand order repaired and checked by reading, not compiled, not run on hardware
//
#include <hip/hip_runtime.h>
#include <math.h>

// CDNA5 / gfx1250, wave32. bf16 WMMA GEMM, fp32 accumulate.
typedef __bf16 bf16_t;
typedef __attribute__((ext_vector_type(16))) __bf16 v16bf;
typedef __attribute__((ext_vector_type(8)))  float  v8f;

#define L_N 4
#define B_N 128
#define H_N 4096
#define BK  64            // K elements staged per LDS tile (2 WMMA k-chunks)
#define PAD 8             // bf16 pad per LDS row -> 144B row stride, conflict-free b128 reads
#define LDW (BK + PAD)    // 72 bf16 per LDS row
#define BN  64            // output columns per block
#define NT  128           // threads per block (4 waves)

union Frag16 { uint4 q[2]; v16bf v; };   // 16 x bf16 = 32B = 2 x uint4

// ---- prologue: states fp32 -> bf16 (4 MB into d_ws; L2-resident thereafter) ----
__global__ __launch_bounds__(256) void cvt_f32_to_bf16_kernel(
    const float* __restrict__ x, bf16_t* __restrict__ y, int n) {
  int i = (blockIdx.x * blockDim.x + threadIdx.x) * 8;
  if (i + 8 <= n) {
    float4 a = ((const float4*)(x + i))[0];
    float4 b = ((const float4*)(x + i))[1];
    union { bf16_t h[8]; uint4 q; } u;
    u.h[0] = (bf16_t)a.x; u.h[1] = (bf16_t)a.y; u.h[2] = (bf16_t)a.z; u.h[3] = (bf16_t)a.w;
    u.h[4] = (bf16_t)b.x; u.h[5] = (bf16_t)b.y; u.h[6] = (bf16_t)b.z; u.h[7] = (bf16_t)b.w;
    ((uint4*)(y + i))[0] = u.q;
  }
}

// ---- main kernel: one block = full batch (M=128) x BN=64 outputs for target t.
// W streamed from HBM exactly once, converted fp32->bf16 in registers.
__global__ __launch_bounds__(NT) void net_wmma_kernel(
    const bf16_t* __restrict__ Abf,   // [L][B][H] bf16 states (scratch)
    const float*  __restrict__ W,     // [L][L][H][H] fp32
    const float*  __restrict__ extv,  // [L][B][H] fp32
    float*        __restrict__ out)   // [L][B][H] fp32
{
  __shared__ __align__(16) bf16_t Asm[B_N * LDW];  // 18 KB A tile (padded rows)

  const int t    = blockIdx.y;
  const int o0   = blockIdx.x * BN;
  const int tid  = threadIdx.x;
  const int wave = tid >> 5;          // 0..3, owns N strip [o0+16w, o0+16w+16)
  const int lane = tid & 31;
  const int ln   = lane & 15;         // N index within strip / M row within tile
  const int lh   = lane >> 4;         // K half selector per WMMA layout
  const int oc   = o0 + wave * 16 + ln;   // this lane's output column == W row

  v8f acc[8] = {};                    // 8 M-tiles x 16x16 fp32 accumulators

  for (int s = 0; s < L_N; ++s) {
    const bf16_t* As   = Abf + (size_t)s * (B_N * H_N);
    const bf16_t* Arow = As + (size_t)tid * H_N;                         // staging row
    const float*  Wrow = W + ((((size_t)t * L_N + s) * H_N) + oc) * H_N; // row oc of W[t][s]

    for (int k0 = 0; k0 < H_N; k0 += BK) {
      // -- cooperative stage: A tile [128 x 64] bf16 -> LDS (128B contig per thread) --
      {
        const uint4* g  = (const uint4*)(Arow + k0);
        uint4* smv = (uint4*)(Asm + tid * LDW);
#pragma unroll
        for (int j = 0; j < 8; ++j) smv[j] = g[j];
      }
      __syncthreads();

      // hint the W stream ahead (global_prefetch_b8)
      __builtin_prefetch(Wrow + k0 + 4 * BK, 0, 1);

#pragma unroll
      for (int kc = 0; kc < 2; ++kc) {
        // -- B fragment: W row oc, K = k0 + kc*32 + lh*16 .. +15 ; fp32->bf16 in-reg --
        Frag16 bf;
        {
          const float4* wp = (const float4*)(Wrow + k0 + kc * 32 + lh * 16);
#pragma unroll
          for (int j = 0; j < 4; ++j) {
            float4 f = wp[j];
            bf.v[4 * j + 0] = (bf16_t)f.x;
            bf.v[4 * j + 1] = (bf16_t)f.y;
            bf.v[4 * j + 2] = (bf16_t)f.z;
            bf.v[4 * j + 3] = (bf16_t)f.w;
          }
        }

        // -- 8 WMMAs: one per 16-row M tile, shared B fragment --
#pragma unroll
        for (int mt = 0; mt < 8; ++mt) {
          Frag16 af;
          const uint4* lp = (const uint4*)(Asm + (mt * 16 + ln) * LDW + kc * 32);
          af.q[0] = lp[2 * lh];       // K = kc*32 + lh*16 .. +7
          af.q[1] = lp[2 * lh + 1];   // K = kc*32 + lh*16 + 8 .. +15
          acc[mt] = __builtin_amdgcn_wmma_f32_16x16x32_bf16(
              false, af.v, false, bf.v, (short)0, acc[mt], false, false);
        }
      }
      __syncthreads();
    }
  }

  // -- epilogue: add ext, tanh, store fp32 per C/D layout (m = mt*16 + lh*8 + r) --
  const float* extb = extv + (size_t)t * (B_N * H_N);
  float*       outb = out  + (size_t)t * (B_N * H_N);
#pragma unroll
  for (int mt = 0; mt < 8; ++mt) {
#pragma unroll
    for (int r = 0; r < 8; ++r) {
      int m = mt * 16 + lh * 8 + r;
      size_t idx = (size_t)m * H_N + oc;
      outb[idx] = tanhf(acc[mt][r] + extb[idx]);
    }
  }
}

extern "C" void kernel_launch(void* const* d_in, const int* in_sizes, int n_in,
                              void* d_out, int out_size, void* d_ws, size_t ws_size,
                              hipStream_t stream) {
  (void)in_sizes; (void)n_in; (void)out_size; (void)ws_size;
  const float* states = (const float*)d_in[0];
  const float* W      = (const float*)d_in[1];
  const float* extv   = (const float*)d_in[2];
  float*  out = (float*)d_out;
  bf16_t* Abf = (bf16_t*)d_ws;                    // L*B*H bf16 = 4 MB scratch

  const int n = L_N * B_N * H_N;                  // 2,097,152
  cvt_f32_to_bf16_kernel<<<n / (256 * 8), 256, 0, stream>>>(states, Abf, n);

  dim3 grid(H_N / BN, L_N);                       // 64 x 4 = 256 blocks
  net_wmma_kernel<<<grid, NT, 0, stream>>>(Abf, W, extv, out);
}
